// NGCF_24550033064402
// MI455X (gfx1250) — compile-verified, matrix operand order checked by reading, not run on hardware
//
#include <hip/hip_runtime.h>
#include <hip/hip_bf16.h>

typedef __attribute__((ext_vector_type(16))) __bf16 v16bf;
typedef __attribute__((ext_vector_type(8)))  float  v8f;

#define USER_NUM 100000
#define ITEM_NUM 50000
#define NNODE    150000
#define EMB      128
#define NNZ_     640000
#define BATCH_   16384
#define NTILES   (NNODE / 16)   // 9375, exact

// ---------- WMMA helpers ----------
__device__ __forceinline__ v8f wmma_bf16(v16bf a, v16bf b, v8f c) {
    // D = A(16x32 bf16) x B(32x16 bf16) + C(16x16 f32)
    return __builtin_amdgcn_wmma_f32_16x16x32_bf16(
        /*neg_a=*/false, a, /*neg_b=*/false, b,
        /*c_mod=*/(short)0, c, /*reuse_a=*/false, /*reuse_b=*/false);
}

// Build an A/B fragment from f32 memory: elements {p[0..7], p[16..23]}
__device__ __forceinline__ v16bf cvt_frag_f32(const float* __restrict__ p) {
    v16bf a;
#pragma unroll
    for (int i = 0; i < 8; ++i) {
        a[i]     = (__bf16)p[i];
        a[8 + i] = (__bf16)p[16 + i];
    }
    return a;
}

// Build a fragment from packed bf16 memory (global or LDS): {p[0..7], p[16..23]}
__device__ __forceinline__ v16bf ld_frag_bf16(const __bf16* __restrict__ p) {
    struct Q { uint4 lo, hi; } q;
    q.lo = *(const uint4*)(p);
    q.hi = *(const uint4*)(p + 16);
    return __builtin_bit_cast(v16bf, q);
}

// ---------- f32 -> bf16 weight conversion ----------
__global__ void k_cvt_bf16(const float* __restrict__ src, __bf16* __restrict__ dst, int n) {
    int i = blockIdx.x * blockDim.x + threadIdx.x;
    if (i < n) dst[i] = (__bf16)src[i];
}

// ---------- agg1 = f_prev (folds the +I term), agg2 = 0 ----------
__global__ void k_prep(const float* __restrict__ su, const float* __restrict__ si, int split,
                       float* __restrict__ agg1, float* __restrict__ agg2, int nrows) {
    int t    = blockIdx.x * blockDim.x + threadIdx.x;
    int row  = t >> 5;
    int lane = t & 31;
    if (row >= nrows) return;
    const float* src = (row < split) ? su + (size_t)row * EMB
                                     : si + (size_t)(row - split) * EMB;
    float4 x = *(const float4*)(src + lane * 4);
    *(float4*)(agg1 + (size_t)row * EMB + lane * 4) = x;
    float4 z = {0.f, 0.f, 0.f, 0.f};
    *(float4*)(agg2 + (size_t)row * EMB + lane * 4) = z;
}

// ---------- COO SpMM: agg1 += v * f[col], agg2 += v * f[col]^2, wave-per-edge ----------
__global__ void k_spmm(const int* __restrict__ rows, const int* __restrict__ cols,
                       const float* __restrict__ vals,
                       const float* __restrict__ su, const float* __restrict__ si, int split,
                       float* __restrict__ agg1, float* __restrict__ agg2, int nnz) {
    int e    = (blockIdx.x * blockDim.x + threadIdx.x) >> 5;
    int lane = threadIdx.x & 31;
    if (e >= nnz) return;
    int   r = rows[e];
    int   c = cols[e];
    float v = vals[e];
    const float* src = (c < split) ? su + (size_t)c * EMB
                                   : si + (size_t)(c - split) * EMB;
    float4 x = *(const float4*)(src + lane * 4);
    size_t o = (size_t)r * EMB + lane * 4;
    atomicAdd(agg1 + o + 0, v * x.x);
    atomicAdd(agg1 + o + 1, v * x.y);
    atomicAdd(agg1 + o + 2, v * x.z);
    atomicAdd(agg1 + o + 3, v * x.w);
    atomicAdd(agg2 + o + 0, v * x.x * x.x);
    atomicAdd(agg2 + o + 1, v * x.y * x.y);
    atomicAdd(agg2 + o + 2, v * x.z * x.z);
    atomicAdd(agg2 + o + 3, v * x.w * x.w);
}

// ---------- fused dual GEMM + bias + leaky_relu ----------
// fout = leaky( agg1 @ W^T + b1 + agg2 @ IW^T + b2 ),  W/IW staged bf16 in LDS
__global__ void __launch_bounds__(128)
k_gnn_gemm(const float* __restrict__ agg1, const float* __restrict__ agg2,
           const __bf16* __restrict__ Wbf, const __bf16* __restrict__ IWbf,
           const float* __restrict__ bias1, const float* __restrict__ bias2,
           float* __restrict__ fout, int ntiles) {
    __shared__ __align__(16) __bf16 sW[EMB * EMB];
    __shared__ __align__(16) __bf16 sIW[EMB * EMB];
    {
        const uint4* g1 = (const uint4*)Wbf;
        const uint4* g2 = (const uint4*)IWbf;
        uint4* s1 = (uint4*)sW;
        uint4* s2 = (uint4*)sIW;
        for (int i = threadIdx.x; i < (EMB * EMB) / 8; i += 128) {
            s1[i] = g1[i];
            s2[i] = g2[i];
        }
    }
    __syncthreads();

    int wave = threadIdx.x >> 5;
    int lane = threadIdx.x & 31;
    int tile = blockIdx.x * 4 + wave;
    if (tile >= ntiles) return;

    int row0 = tile * 16;
    int mrow = row0 + (lane & 15);
    int hi   = (lane >> 4) ? 8 : 0;

    v8f acc[8];
    v8f z = {0.f, 0.f, 0.f, 0.f, 0.f, 0.f, 0.f, 0.f};
#pragma unroll
    for (int nt = 0; nt < 8; ++nt) acc[nt] = z;

#pragma unroll
    for (int kc = 0; kc < 4; ++kc) {
        int   kbase = kc * 32 + hi;
        v16bf a1 = cvt_frag_f32(agg1 + (size_t)mrow * EMB + kbase);
        v16bf a2 = cvt_frag_f32(agg2 + (size_t)mrow * EMB + kbase);
#pragma unroll
        for (int nt = 0; nt < 8; ++nt) {
            int   j  = nt * 16 + (lane & 15);
            v16bf b1 = ld_frag_bf16(sW  + j * EMB + kbase);
            v16bf b2 = ld_frag_bf16(sIW + j * EMB + kbase);
            acc[nt] = wmma_bf16(a1, b1, acc[nt]);
            acc[nt] = wmma_bf16(a2, b2, acc[nt]);
        }
    }

#pragma unroll
    for (int nt = 0; nt < 8; ++nt) {
        int   j  = nt * 16 + (lane & 15);
        float bb = bias1[j] + bias2[j];
#pragma unroll
        for (int r = 0; r < 8; ++r) {
            int   row = row0 + r + hi;
            float x   = acc[nt][r] + bb;
            fout[(size_t)row * EMB + j] = (x > 0.f) ? x : 0.01f * x;
        }
    }
}

// ---------- gather final_embd segments into contiguous [B,1024] ----------
__global__ void k_gather(const int* __restrict__ uidx, const int* __restrict__ iidx,
                         const float* __restrict__ u_embd, const float* __restrict__ i_embd,
                         const float* __restrict__ f1, const float* __restrict__ f2,
                         const float* __restrict__ f3, float* __restrict__ embd) {
    int b   = blockIdx.x;
    int s   = threadIdx.x;        // 256 float4 slots = 1024 floats
    int seg = s >> 5;             // 0..7 (each segment = 128 floats = 32 slots)
    int off = (s & 31) * 4;
    int u  = uidx[b];
    int it = iidx[b];
    const float* base = u_embd;
    switch (seg) {
        case 0: base = u_embd + (size_t)u * EMB; break;
        case 1: base = f1 + (size_t)u * EMB; break;
        case 2: base = f2 + (size_t)u * EMB; break;
        case 3: base = f3 + (size_t)u * EMB; break;
        case 4: base = i_embd + (size_t)it * EMB; break;
        case 5: base = f1 + (size_t)(it + USER_NUM) * EMB; break;
        case 6: base = f2 + (size_t)(it + USER_NUM) * EMB; break;
        default: base = f3 + (size_t)(it + USER_NUM) * EMB; break;
    }
    float4 x = *(const float4*)(base + off);
    *(float4*)(embd + (size_t)b * 1024 + s * 4) = x;
}

// ---------- h1 = relu(embd @ t1_w^T + t1_b) via WMMA, K=1024 ----------
__global__ void __launch_bounds__(128)
k_mlp1(const float* __restrict__ embd, const __bf16* __restrict__ t1wbf,
       const float* __restrict__ t1b, float* __restrict__ h1) {
    int wave = threadIdx.x >> 5;
    int lane = threadIdx.x & 31;
    int row0 = (blockIdx.x * 4 + wave) * 16;
    int mrow = row0 + (lane & 15);
    int hi   = (lane >> 4) ? 8 : 0;

    v8f acc[4];
    v8f z = {0.f, 0.f, 0.f, 0.f, 0.f, 0.f, 0.f, 0.f};
#pragma unroll
    for (int nt = 0; nt < 4; ++nt) acc[nt] = z;

    for (int kc = 0; kc < 32; ++kc) {
        int   kbase = kc * 32 + hi;
        v16bf a = cvt_frag_f32(embd + (size_t)mrow * 1024 + kbase);
#pragma unroll
        for (int nt = 0; nt < 4; ++nt) {
            int   j = nt * 16 + (lane & 15);
            v16bf b = ld_frag_bf16(t1wbf + (size_t)j * 1024 + kbase);
            acc[nt] = wmma_bf16(a, b, acc[nt]);
        }
    }

#pragma unroll
    for (int nt = 0; nt < 4; ++nt) {
        int   j  = nt * 16 + (lane & 15);
        float bb = t1b[j];
#pragma unroll
        for (int r = 0; r < 8; ++r) {
            int   row = row0 + r + hi;
            float x   = acc[nt][r] + bb;
            h1[(size_t)row * 64 + j] = (x > 0.f) ? x : 0.f;
        }
    }
}

// ---------- tail: h2 = relu(h1 @ t2_w^T + t2_b); out = h2 @ t3_w^T + t3_b ----------
__global__ void k_mlp_tail(const float* __restrict__ h1,
                           const float* __restrict__ t2w, const float* __restrict__ t2b,
                           const float* __restrict__ t3w, const float* __restrict__ t3b,
                           float* __restrict__ out) {
    __shared__ float sh[8][64];
    int wave = threadIdx.x >> 5;
    int lane = threadIdx.x & 31;
    int b    = blockIdx.x * 8 + wave;
    sh[wave][lane]      = h1[(size_t)b * 64 + lane];
    sh[wave][lane + 32] = h1[(size_t)b * 64 + lane + 32];
    __syncthreads();
    float h2 = t2b[lane];
    const float* wrow = t2w + (size_t)lane * 64;
#pragma unroll
    for (int k = 0; k < 64; ++k) h2 += sh[wave][k] * wrow[k];
    h2 = (h2 > 0.f) ? h2 : 0.f;
    float partial = h2 * t3w[lane];
#pragma unroll
    for (int m = 16; m; m >>= 1) partial += __shfl_xor(partial, m, 32);
    if (lane == 0) out[b] = partial + t3b[0];
}

extern "C" void kernel_launch(void* const* d_in, const int* in_sizes, int n_in,
                              void* d_out, int out_size, void* d_ws, size_t ws_size,
                              hipStream_t stream) {
    const int*   uidx   = (const int*)d_in[0];
    const int*   iidx   = (const int*)d_in[1];
    const int*   rows   = (const int*)d_in[2];
    const int*   cols   = (const int*)d_in[3];
    const float* vals   = (const float*)d_in[4];
    const float* u_embd = (const float*)d_in[5];
    const float* i_embd = (const float*)d_in[6];
    const float* gnn_w  = (const float*)d_in[7];
    const float* gnn_b  = (const float*)d_in[8];
    const float* gnn_iw = (const float*)d_in[9];
    const float* gnn_ib = (const float*)d_in[10];
    const float* t1_w   = (const float*)d_in[11];
    const float* t1_b   = (const float*)d_in[12];
    const float* t2_w   = (const float*)d_in[13];
    const float* t2_b   = (const float*)d_in[14];
    const float* t3_w   = (const float*)d_in[15];
    const float* t3_b   = (const float*)d_in[16];

    char*        ws = (char*)d_ws;
    const size_t NF = (size_t)NNODE * EMB * sizeof(float);   // 76.8 MB

    float* fbuf[3] = { (float*)(ws + 0 * NF), (float*)(ws + 1 * NF), (float*)(ws + 2 * NF) };
    float* agg1    = (float*)(ws + 3 * NF);
    float* agg2    = (float*)(ws + 4 * NF);
    // aliases (used only after the GNN loop, when agg buffers are dead):
    float* embd = agg1;                          // BATCH*1024*4 = 67 MB  <= 76.8 MB
    float* h1   = agg2;                          // BATCH*64*4   = 4 MB
    __bf16* Wbf   = (__bf16*)(ws + 5 * NF);
    __bf16* IWbf  = (__bf16*)(ws + 5 * NF + 32768);
    __bf16* t1wbf = (__bf16*)(ws + 5 * NF + 65536);

    const int prep_blocks = (NNODE * 32 + 255) / 256;   // 18750
    const int spmm_blocks = (NNZ_ * 32 + 255) / 256;    // 80000
    const int gemm_blocks = (NTILES + 3) / 4;           // 2344

    for (int l = 0; l < 3; ++l) {
        const float* pu    = (l == 0) ? u_embd : fbuf[l - 1];
        const float* pi    = (l == 0) ? i_embd : fbuf[l - 1];
        int          split = (l == 0) ? USER_NUM : NNODE;

        k_cvt_bf16<<<64, 256, 0, stream>>>(gnn_w  + (size_t)l * EMB * EMB, Wbf,  EMB * EMB);
        k_cvt_bf16<<<64, 256, 0, stream>>>(gnn_iw + (size_t)l * EMB * EMB, IWbf, EMB * EMB);
        k_prep<<<prep_blocks, 256, 0, stream>>>(pu, pi, split, agg1, agg2, NNODE);
        k_spmm<<<spmm_blocks, 256, 0, stream>>>(rows, cols, vals, pu, pi, split,
                                                agg1, agg2, NNZ_);
        k_gnn_gemm<<<gemm_blocks, 128, 0, stream>>>(agg1, agg2, Wbf, IWbf,
                                                    gnn_b + l * EMB, gnn_ib + l * EMB,
                                                    fbuf[l], NTILES);
    }

    k_cvt_bf16<<<256, 256, 0, stream>>>(t1_w, t1wbf, 64 * 1024);
    k_gather<<<BATCH_, 256, 0, stream>>>(uidx, iidx, u_embd, i_embd,
                                         fbuf[0], fbuf[1], fbuf[2], embd);
    k_mlp1<<<BATCH_ / 64, 128, 0, stream>>>(embd, t1wbf, t1_b, h1);
    k_mlp_tail<<<BATCH_ / 8, 256, 0, stream>>>(h1, t2_w, t2_b, t3_w, t3_b, (float*)d_out);
}
